// SynthesizerAttention_18476949307518
// MI455X (gfx1250) — matrix operand order repaired and checked by reading, not compiled, not run on hardware
//
#include <hip/hip_runtime.h>

// ---------------------------------------------------------------------------
// Synthesizer attention, fused flash-style for MI455X (gfx1250, wave32 WMMA).
//   B=2, T=2048, C=1024, H=16, D=64
// Pipeline:
//   1) f32->f16 conversions (x, W1, Wv, Wp) + fused cvt+transpose of w2
//   2) h  = relu(x·W1^T + b1)   [WMMA GEMM 16x64/wave, f16 row-major out]
//   3) vT = (x·Wv^T + bv)^T     [same GEMM, transposed f16 out: [B,H,64,T]]
//   4) fused causal softmax attention: S^T/O^T WMMA formulation, online
//      softmax; all fragments loaded as contiguous b128 vectors
//   5) out = y·Wp^T + bp        [WMMA GEMM, f32 out]
// Scores tensor (512 MB in f32) never touches HBM; all matmuls on
// v_wmma_f32_16x16x32_f16 (f32 accumulate).
// ---------------------------------------------------------------------------

typedef __attribute__((ext_vector_type(16))) _Float16 v16h;
typedef __attribute__((ext_vector_type(8)))  _Float16 v8h;
typedef __attribute__((ext_vector_type(8)))  float    v8f;

__device__ __forceinline__ v8f wmma_f16(v16h a, v16h b, v8f c) {
  // (neg_a, A, neg_b, B, c_mod, C, reuse_a, reuse_b)
  return __builtin_amdgcn_wmma_f32_16x16x32_f16(false, a, false, b, (short)0, c,
                                                false, false);
}

// A-fragment (16x32 f16) from a [rows x K] row-major source where the lane's
// row is fixed and K is contiguous: halves 0..15 = K base+g*16.. (contiguous).
__device__ __forceinline__ v16h load_afrag(const _Float16* rowp, int g) {
  v8h lo = *(const v8h*)(rowp + g * 16);
  v8h hi = *(const v8h*)(rowp + g * 16 + 8);
  return __builtin_shufflevector(lo, hi, 0, 1, 2, 3, 4, 5, 6, 7,
                                 8, 9, 10, 11, 12, 13, 14, 15);
}

__global__ void cvt_f32_f16(const float* __restrict__ in,
                            _Float16* __restrict__ out, int n) {
  int i = blockIdx.x * blockDim.x + threadIdx.x;
  if (i < n) out[i] = (_Float16)in[i];
}

// w2 [64][2048] f32 -> w2T [2048][64] f16 (fused convert + transpose)
__global__ void cvt_tr_w2(const float* __restrict__ in,
                          _Float16* __restrict__ out) {
  int i = blockIdx.x * blockDim.x + threadIdx.x;   // over 64*2048
  int d = i >> 11;
  int j = i & 2047;
  out[j * 64 + d] = (_Float16)in[i];
}

// out[m][n] = act( sum_k A[m][k] * W[n][k] + bias[n] )
// A: [M,K] f16 row-major; W: [N,K] f16 row-major (torch Linear weight).
// One wave per 16x64 output tile (A-frag reused across 4 N-tiles).
// out_mode: 0 = f32 row-major, 1 = f16 row-major,
//           2 = f16 head-transposed: out[((row/T)*H + col/64)*64 + col%64][T] + row%T
__global__ void gemm_xwt(const _Float16* __restrict__ A,
                         const _Float16* __restrict__ W,
                         const float* __restrict__ bias,
                         void* __restrict__ out,
                         int M, int N, int K, int Tdim,
                         int do_relu, int out_mode) {
  int wave = (int)((blockIdx.x * blockDim.x + threadIdx.x) >> 5);
  int lane = threadIdx.x & 31;
  int tilesN = N >> 6;                      // 64-wide N tiles
  int tm = (wave / tilesN) << 4;
  int tn = (wave % tilesN) << 6;
  if (tm >= M) return;                      // wave-uniform
  int tq = lane & 15;
  int g  = lane >> 4;

  const _Float16* arow = A + (size_t)(tm + tq) * K;
  const _Float16* w0 = W + (size_t)(tn + tq) * K;
  const _Float16* w1 = W + (size_t)(tn + 16 + tq) * K;
  const _Float16* w2p = W + (size_t)(tn + 32 + tq) * K;
  const _Float16* w3 = W + (size_t)(tn + 48 + tq) * K;

  v8f ac0 = {0,0,0,0,0,0,0,0}, ac1 = ac0, ac2 = ac0, ac3 = ac0;
  for (int k0 = 0; k0 < K; k0 += 32) {
    v16h af = load_afrag(arow + k0, g);
    // B frag (32x16 f16): lane = col N, halves = K g*16..g*16+15 (contiguous)
    v16h b0 = *(const v16h*)(w0 + k0 + g * 16);
    v16h b1 = *(const v16h*)(w1 + k0 + g * 16);
    v16h b2 = *(const v16h*)(w2p + k0 + g * 16);
    v16h b3 = *(const v16h*)(w3 + k0 + g * 16);
    ac0 = wmma_f16(af, b0, ac0);
    ac1 = wmma_f16(af, b1, ac1);
    ac2 = wmma_f16(af, b2, ac2);
    ac3 = wmma_f16(af, b3, ac3);
  }

  int heads = N >> 6;
#pragma unroll
  for (int ct = 0; ct < 4; ++ct) {
    int col = tn + ct * 16 + tq;
    float bb = bias[col];
    v8f acc = ct == 0 ? ac0 : ct == 1 ? ac1 : ct == 2 ? ac2 : ac3;
#pragma unroll
    for (int r = 0; r < 8; ++r) {
      int row = tm + g * 8 + r;             // C/D frag: M = 8*g + r
      float v = acc[r] + bb;
      if (do_relu) v = fmaxf(v, 0.f);
      if (out_mode == 0) {
        ((float*)out)[(size_t)row * N + col] = v;
      } else if (out_mode == 1) {
        ((_Float16*)out)[(size_t)row * N + col] = (_Float16)v;
      } else {
        int b = row / Tdim, t = row % Tdim;
        size_t idx = (((size_t)b * heads + (col >> 6)) * 64 + (col & 63));
        ((_Float16*)out)[idx * Tdim + t] = (_Float16)v;
      }
    }
  }
}

// Fused causal synthesizer attention. One wave per (b, head, 16-query tile).
// S^T = A(w2T rows) x B(h)   -> lane carries query t, VGPR carries key j
// O^T = A(vT rows)  x B(p^T) -> lane carries query t, VGPR carries dim d
__global__ void synth_attn(const _Float16* __restrict__ Hm,   // [B*T, C] f16
                           const _Float16* __restrict__ VT,   // [B,H,64,T] f16
                           const _Float16* __restrict__ W2T,  // [T, 64] f16
                           const float* __restrict__ B2,      // [T] f32
                           _Float16* __restrict__ Y) {        // [B*T, C] f16
  const int T = 2048, C = 1024, Dh = 64;
  int wave = (int)((blockIdx.x * blockDim.x + threadIdx.x) >> 5);
  int lane = threadIdx.x & 31;
  int qt   = wave & 127;          // T/16 = 128 query tiles
  int head = (wave >> 7) & 15;
  int b    = wave >> 11;
  int q0   = qt << 4;
  int tq   = lane & 15;
  int g    = lane >> 4;
  int tglob = q0 + tq;            // this lane's query row

  // h B-frags (loop invariant): B[k=d][n=t], lane=t, halves = d contiguous
  const _Float16* hrow = Hm + ((size_t)(b * T + tglob)) * C + head * Dh;
  v16h bq0 = *(const v16h*)(hrow + g * 16);        // d = 0..31 phase
  v16h bq1 = *(const v16h*)(hrow + 32 + g * 16);   // d = 32..63 phase

  // vT row pointers (loop invariant): A[m=d][k=j] with j contiguous
  const _Float16* vtbase = VT + ((size_t)(b * 16 + head) * 64) * T;
  const _Float16* vr0 = vtbase + (size_t)(0 * 16 + tq) * T;
  const _Float16* vr1 = vtbase + (size_t)(1 * 16 + tq) * T;
  const _Float16* vr2 = vtbase + (size_t)(2 * 16 + tq) * T;
  const _Float16* vr3 = vtbase + (size_t)(3 * 16 + tq) * T;

  float m_run = -3.0e38f, l_run = 0.f;
  v8f o0 = {0,0,0,0,0,0,0,0}, o1 = o0, o2 = o0, o3 = o0;

  for (int j0 = 0; j0 <= q0 + 15; j0 += 32) {      // wave-uniform bound
    // prefetch next key block of V rows (gfx1250 global_prefetch_b8)
    if (j0 + 32 <= q0 + 15) {
      __builtin_prefetch(vr0 + j0 + 32, 0, 1);
      __builtin_prefetch(vr2 + j0 + 32, 0, 1);
    }

    float p[16];
#pragma unroll
    for (int jf = 0; jf < 2; ++jf) {               // two 16-key M tiles
      // A frag from w2T: lane row j = j0+jf*16+tq, K = d contiguous
      const _Float16* w2row = W2T + (size_t)(j0 + jf * 16 + tq) * Dh;
      v8f sacc = {0,0,0,0,0,0,0,0};
      sacc = wmma_f16(load_afrag(w2row, g), bq0, sacc);        // d 0..31
      sacc = wmma_f16(load_afrag(w2row + 32, g), bq1, sacc);   // d 32..63
      // bias + causal mask; s[j][t]: lane j-range = jf*16 + 8g + r
      v8f bvec = *(const v8f*)(B2 + j0 + jf * 16 + g * 8);
#pragma unroll
      for (int r = 0; r < 8; ++r) {
        int jj = j0 + jf * 16 + g * 8 + r;
        float s = sacc[r] + bvec[r];
        if (jj > tglob) s = -1.0e10f;              // causal mask
        p[jf * 8 + r] = s;
      }
    }

    // online softmax stats for query tq (lanes l and l+16 keep equal copies)
    float bm = p[0];
#pragma unroll
    for (int i = 1; i < 16; ++i) bm = fmaxf(bm, p[i]);
    bm = fmaxf(bm, __shfl_xor(bm, 16, 32));
    float m_new = fmaxf(m_run, bm);
    float scale = __expf(m_run - m_new);
    float ls = 0.f;
#pragma unroll
    for (int i = 0; i < 16; ++i) { p[i] = __expf(p[i] - m_new); ls += p[i]; }
    ls += __shfl_xor(ls, 16, 32);
    l_run = l_run * scale + ls;
    m_run = m_new;
#pragma unroll
    for (int r = 0; r < 8; ++r) {
      o0[r] *= scale; o1[r] *= scale; o2[r] *= scale; o3[r] *= scale;
    }

    // Build P^T B-frag (K = 32 keys): lane needs p[t=lane%16][j0 + g*16 + i]
    v16h pb;
#pragma unroll
    for (int r = 0; r < 8; ++r) {
      float x0 = p[r], x1 = p[8 + r];
      float y0 = __shfl_xor(x0, 16, 32);
      float y1 = __shfl_xor(x1, 16, 32);
      pb[r]     = (_Float16)(g == 0 ? x0 : y1);
      pb[r + 8] = (_Float16)(g == 0 ? y0 : x1);
    }

    // O^T += V^T x P^T : A frag from vT rows (lane row d, K = j contiguous)
    o0 = wmma_f16(load_afrag(vr0 + j0, g), pb, o0);
    o1 = wmma_f16(load_afrag(vr1 + j0, g), pb, o1);
    o2 = wmma_f16(load_afrag(vr2 + j0, g), pb, o2);
    o3 = wmma_f16(load_afrag(vr3 + j0, g), pb, o3);
  }

  // y[b, t, head*64 + d] = o / l ; O^T frag: lane = t, d = dt*16 + 8g + r
  float inv = 1.0f / l_run;
  _Float16* yrow = Y + ((size_t)(b * T + tglob)) * C + head * Dh;
#pragma unroll
  for (int r = 0; r < 8; ++r) {
    yrow[0 * 16 + g * 8 + r] = (_Float16)(o0[r] * inv);
    yrow[1 * 16 + g * 8 + r] = (_Float16)(o1[r] * inv);
    yrow[2 * 16 + g * 8 + r] = (_Float16)(o2[r] * inv);
    yrow[3 * 16 + g * 8 + r] = (_Float16)(o3[r] * inv);
  }
}

extern "C" void kernel_launch(void* const* d_in, const int* in_sizes, int n_in,
                              void* d_out, int out_size, void* d_ws, size_t ws_size,
                              hipStream_t stream) {
  (void)in_sizes; (void)n_in; (void)out_size; (void)ws_size;
  const int B = 2, T = 2048, C = 1024;
  const size_t BT = (size_t)B * T;

  const float* x   = (const float*)d_in[0];
  const float* w1w = (const float*)d_in[1];
  const float* w1b = (const float*)d_in[2];
  const float* w2  = (const float*)d_in[3];
  const float* b2  = (const float*)d_in[4];
  const float* vw  = (const float*)d_in[5];
  const float* vb  = (const float*)d_in[6];
  const float* pw  = (const float*)d_in[7];
  const float* pbs = (const float*)d_in[8];
  float* out = (float*)d_out;

  char* ws = (char*)d_ws;
  size_t off = 0;
  auto take = [&](size_t elems) -> _Float16* {
    _Float16* p = (_Float16*)(ws + off);
    off += ((elems * sizeof(_Float16)) + 255) & ~(size_t)255;
    return p;
  };
  _Float16* xh   = take(BT * C);
  _Float16* w1h  = take((size_t)C * C);
  _Float16* vwh  = take((size_t)C * C);
  _Float16* pwh  = take((size_t)C * C);
  _Float16* w2th = take((size_t)T * 64);          // w2^T : [T][64]
  _Float16* hh   = take(BT * C);                  // h    : [B*T][C]
  _Float16* vth  = take(BT * C);                  // v^T  : [B][H][64][T]
  _Float16* yh   = take(BT * C);                  // y    : [B*T][C]

  auto cvt = [&](const float* src, _Float16* dst, size_t n) {
    int nb = (int)((n + 255) / 256);
    cvt_f32_f16<<<dim3(nb), dim3(256), 0, stream>>>(src, dst, (int)n);
  };
  cvt(x,   xh,  BT * C);
  cvt(w1w, w1h, (size_t)C * C);
  cvt(vw,  vwh, (size_t)C * C);
  cvt(pw,  pwh, (size_t)C * C);
  cvt_tr_w2<<<dim3((64 * T) / 256), dim3(256), 0, stream>>>(w2, w2th);

  const int M = (int)BT, N = C, K = C;
  const int waves = (M / 16) * (N / 64);            // 4096 waves, 16x64 tiles
  dim3 ggrid(waves / 8), gblock(256);               // 8 waves / block
  gemm_xwt<<<ggrid, gblock, 0, stream>>>(xh, w1h, w1b, hh,  M, N, K, T, 1, 1);
  gemm_xwt<<<ggrid, gblock, 0, stream>>>(xh, vwh, vb,  vth, M, N, K, T, 0, 2);

  // B*H*(T/16) = 4096 waves -> 512 blocks of 256 threads
  synth_attn<<<dim3(512), dim3(256), 0, stream>>>(hh, vth, w2th, b2, yh);

  gemm_xwt<<<ggrid, gblock, 0, stream>>>(yh, pwh, pbs, out, M, N, K, T, 0, 0);
}
